// Attention_10015863734775
// MI455X (gfx1250) — matrix operand order repaired and checked by reading, not compiled, not run on hardware
//
#include <hip/hip_runtime.h>
#include <hip/hip_bf16.h>
#include <math.h>

// ---------------------------------------------------------------------------
// MI455X (gfx1250) attention block: QKV proj + RoPE + signed-softmax flash
// attention + output proj, all matmuls on v_wmma_f32_16x16x32_f16 (wave32).
// LDS tiles padded (+16B/row) for conflict-free b128 fragment reads.
// ---------------------------------------------------------------------------

typedef _Float16 h16 __attribute__((ext_vector_type(16)));
typedef _Float16 h8  __attribute__((ext_vector_type(8)));
typedef _Float16 h4  __attribute__((ext_vector_type(4)));
typedef float    f8  __attribute__((ext_vector_type(8)));
typedef float    f4  __attribute__((ext_vector_type(4)));

#define SEQ     2048
#define DMODEL  2048
#define DHEAD   128
#define NHEADS  16

#define GS  40          // GEMM LDS row stride in halves (32 + 8 pad)
#define KS  136         // attention LDS row stride in halves (128 + 8 pad)

__device__ __forceinline__ f8 f8zero() {
    f8 z;
#pragma unroll
    for (int i = 0; i < 8; ++i) z[i] = 0.0f;
    return z;
}

__device__ __forceinline__ f8 wmma16(h16 a, h16 b, f8 c) {
    return __builtin_amdgcn_wmma_f32_16x16x32_f16(
        false, a, false, b, (short)0, c, false, false);
}

union HU { h16 v; h8 h[2]; };

__device__ __forceinline__ void wait_lds() {
#if __has_builtin(__builtin_amdgcn_s_wait_dscnt)
    __builtin_amdgcn_s_wait_dscnt(0);
#else
    asm volatile("s_wait_dscnt 0" ::: "memory");
#endif
}

// ---------------------------------------------------------------------------
// Kernel 1: projection GEMM, templated epilogue.
//   ZM=0 : C = A*W + b  -> f32 [s][head*128+e]            (Q and K)
//   ZM=1 : C = A*W + b  -> f16 transposed [head*128+e][s] (V)
// ---------------------------------------------------------------------------
template <int ZM>
__global__ __launch_bounds__(256) void proj_gemm(
    const float* __restrict__ Ag, const float* __restrict__ W,
    const float* __restrict__ bias,
    float* __restrict__ outf, _Float16* __restrict__ outt)
{
    const int bn = blockIdx.x;   // head (128-col tile)
    const int bm = blockIdx.y;   // 128-row tile

    __shared__ __align__(32) _Float16 Al[128 * GS];
    __shared__ __align__(32) _Float16 Bl[128 * GS];

    const int tid = threadIdx.x;
    const int w   = tid >> 5, lane = tid & 31;
    const int wr  = w >> 1,   wc   = w & 1;
    const int ln  = lane & 15, lh  = lane >> 4;

    f8 acc[2][4];
#pragma unroll
    for (int i = 0; i < 2; ++i)
#pragma unroll
        for (int t = 0; t < 4; ++t) acc[i][t] = f8zero();

    for (int kt = 0; kt < DMODEL / 32; ++kt) {
        const int k0 = kt * 32;
        // stage A tile (128x32 f32 -> f16)
#pragma unroll
        for (int i = 0; i < 4; ++i) {
            int f = tid + 256 * i;          // 1024 float4's
            int row = f >> 3, c4 = (f & 7) << 2;
            f4 v = *(const f4*)(Ag + (size_t)(bm * 128 + row) * DMODEL + k0 + c4);
            h4 hv;
#pragma unroll
            for (int j = 0; j < 4; ++j) hv[j] = (_Float16)v[j];
            *(h4*)(Al + row * GS + c4) = hv;
        }
        // stage B tile: Bl[n][k] = W[bn][k0+k][n]
#pragma unroll
        for (int i = 0; i < 4; ++i) {
            int f = tid + 256 * i;
            int k = f >> 5, n4 = (f & 31) << 2;
            f4 v = *(const f4*)(W + (size_t)bn * DMODEL * DHEAD + (k0 + k) * DHEAD + n4);
#pragma unroll
            for (int j = 0; j < 4; ++j) Bl[(n4 + j) * GS + k] = (_Float16)v[j];
        }
        // prefetch next k-slab while this one computes
        if (kt < DMODEL / 32 - 1) {
            if (tid < 128)
                __builtin_prefetch(Ag + (size_t)(bm * 128 + tid) * DMODEL + k0 + 32, 0, 1);
            else {
                int kr = (tid - 128) >> 2, q4 = (tid - 128) & 3;
                __builtin_prefetch(W + (size_t)bn * DMODEL * DHEAD +
                                   (k0 + 32 + kr) * DHEAD + q4 * 32, 0, 1);
            }
        }
        __syncthreads();

        h16 a[2], b[4];
#pragma unroll
        for (int i = 0; i < 2; ++i) {
            int row = wr * 32 + i * 16 + ln;
            HU u;
            u.h[0] = *(const h8*)(Al + row * GS + 8 * lh);
            u.h[1] = *(const h8*)(Al + row * GS + 16 + 8 * lh);
            a[i] = u.v;
        }
#pragma unroll
        for (int t = 0; t < 4; ++t) {
            int col = wc * 64 + t * 16 + ln;
            HU u;
            u.h[0] = *(const h8*)(Bl + col * GS + 8 * lh);
            u.h[1] = *(const h8*)(Bl + col * GS + 16 + 8 * lh);
            b[t] = u.v;
        }
#pragma unroll
        for (int i = 0; i < 2; ++i)
#pragma unroll
            for (int t = 0; t < 4; ++t)
                acc[i][t] = wmma16(a[i], b[t], acc[i][t]);
        __syncthreads();
    }

#pragma unroll
    for (int i = 0; i < 2; ++i) {
#pragma unroll
        for (int t = 0; t < 4; ++t) {
            int e = wc * 64 + t * 16 + ln;
            float bv2 = bias[bn * DHEAD + e];
            int s0 = bm * 128 + wr * 32 + i * 16 + 8 * lh;
#pragma unroll
            for (int r = 0; r < 8; ++r) {
                float val = acc[i][t][r] + bv2;
                int s = s0 + r;
                if constexpr (ZM == 0)
                    outf[(size_t)s * DMODEL + bn * DHEAD + e] = val;
                else
                    outt[(size_t)(bn * DHEAD + e) * SEQ + s] = (_Float16)val;
            }
        }
    }
}

// ---------------------------------------------------------------------------
// Kernel 2: RoPE on Q and K, f32 -> f16 [s][h*128+d].
// ---------------------------------------------------------------------------
__global__ __launch_bounds__(256) void rope_kernel(
    const float* __restrict__ qraw, const float* __restrict__ kraw,
    _Float16* __restrict__ qh, _Float16* __restrict__ kh)
{
    int idx = blockIdx.x * 256 + threadIdx.x;    // SEQ * 1024 threads
    int s = idx >> 10;
    int p = idx & 1023;
    int h = p >> 6;
    int i = p & 63;
    float inv = __expf(-(float)(2 * i) * (9.210340371976184f / 128.0f));
    float fr  = (float)s * inv;
    float sn, cs;
    sincosf(fr, &sn, &cs);
    int base = s * DMODEL + h * DHEAD + 2 * i;
    float xr = qraw[base], xi = qraw[base + 1];
    qh[base]     = (_Float16)(xr * cs - xi * sn);
    qh[base + 1] = (_Float16)(xr * sn + xi * cs);
    xr = kraw[base]; xi = kraw[base + 1];
    kh[base]     = (_Float16)(xr * cs - xi * sn);
    kh[base + 1] = (_Float16)(xr * sn + xi * cs);
}

// ---------------------------------------------------------------------------
// Kernel 3: flash-style signed-softmax attention.
//   grid=(16 qblocks, 16 heads); 8 waves x 16 query rows.
//   K/V tiles cooperatively staged in (dynamic) LDS once per key block.
//   attn_w = sgn(s)*exp(|s| - rowmax|s|), normalized by sum|attn_w| + 1e-6.
// ---------------------------------------------------------------------------
__global__ __launch_bounds__(256) void attn_kernel(
    const _Float16* __restrict__ qh, const _Float16* __restrict__ khb,
    const _Float16* __restrict__ vt, _Float16* __restrict__ zh)
{
    const int qb   = blockIdx.x;
    const int head = blockIdx.y;
    const int tid  = threadIdx.x;
    const int w    = tid >> 5, lane = tid & 31;
    const int ln   = lane & 15, lh  = lane >> 4;

    extern __shared__ __align__(32) _Float16 smem[];
    _Float16* Kl = smem;                       // [128 keys][KS]  (B frags: [key][d])
    _Float16* Vl = smem + 128 * KS;            // [128 d][KS]     (B frags: [d][key])
    _Float16* Pb = smem + 2 * 128 * KS;        // 8 waves x [16][KS]
    _Float16* pw = Pb + w * 16 * KS;

    // Q A-fragments for this wave's 16 rows (held in VGPRs for the whole loop)
    h16 aq[4];
    {
        const int qrow = qb * 128 + w * 16 + ln;
        const _Float16* qp = qh + (size_t)qrow * DMODEL + head * DHEAD;
#pragma unroll
        for (int c = 0; c < 4; ++c) {
            HU u;
            u.h[0] = *(const h8*)(qp + 32 * c + 8 * lh);
            u.h[1] = *(const h8*)(qp + 32 * c + 16 + 8 * lh);
            aq[c] = u.v;
        }
    }

    f8 zacc[8];
    float M[8], L[8];
#pragma unroll
    for (int t = 0; t < 8; ++t) zacc[t] = f8zero();
#pragma unroll
    for (int r = 0; r < 8; ++r) { M[r] = -3.0e38f; L[r] = 0.0f; }

    const float iscale = 0.0883883476483184f;  // 1/sqrt(128)

    for (int kb = 0; kb <= qb; ++kb) {
        // ---- stage K and V tiles into LDS (all 8 waves cooperate) ----
#pragma unroll
        for (int i = 0; i < 8; ++i) {
            int idx = tid + 256 * i;            // 2048 h8-chunks per tile
            int row = idx >> 4, c8 = (idx & 15) << 3;
            *(h8*)(Kl + row * KS + c8) =
                *(const h8*)(khb + (size_t)(kb * 128 + row) * DMODEL + head * DHEAD + c8);
            *(h8*)(Vl + row * KS + c8) =
                *(const h8*)(vt + (size_t)(head * DHEAD + row) * SEQ + kb * 128 + c8);
        }
        // prefetch next key block while computing this one
        if (kb < qb) {
            if (tid < 128)
                __builtin_prefetch(khb + (size_t)((kb + 1) * 128 + tid) * DMODEL +
                                   head * DHEAD, 0, 1);
            else
                __builtin_prefetch(vt + (size_t)(head * DHEAD + (tid - 128)) * SEQ +
                                   (kb + 1) * 128, 0, 1);
        }
        __syncthreads();

        // ---- scores S = Q * K^T (16 x 128) ----
        f8 sacc[8];
#pragma unroll
        for (int t = 0; t < 8; ++t) sacc[t] = f8zero();
#pragma unroll
        for (int t = 0; t < 8; ++t) {
            const _Float16* kp = Kl + (t * 16 + ln) * KS + 8 * lh;
#pragma unroll
            for (int c = 0; c < 4; ++c) {
                HU u;
                u.h[0] = *(const h8*)(kp + 32 * c);
                u.h[1] = *(const h8*)(kp + 32 * c + 16);
                sacc[t] = wmma16(aq[c], u.v, sacc[t]);
            }
        }

        // ---- scale + causal mask + |s| row max ----
        float rmax[8];
#pragma unroll
        for (int r = 0; r < 8; ++r) rmax[r] = 0.0f;
        const bool diag = (kb == qb);
        const int qrel0 = w * 16 + 8 * lh;
#pragma unroll
        for (int t = 0; t < 8; ++t)
#pragma unroll
            for (int r = 0; r < 8; ++r) {
                float sv = sacc[t][r] * iscale;
                if (diag && (t * 16 + ln) > (qrel0 + r)) sv = 0.0f;
                sacc[t][r] = sv;
                rmax[r] = fmaxf(rmax[r], fabsf(sv));
            }
#pragma unroll
        for (int m = 1; m < 16; m <<= 1)
#pragma unroll
            for (int r = 0; r < 8; ++r)
                rmax[r] = fmaxf(rmax[r], __shfl_xor(rmax[r], m, 32));

        float alpha[8], rsum[8];
#pragma unroll
        for (int r = 0; r < 8; ++r) {
            float Mn = fmaxf(M[r], rmax[r]);
            alpha[r] = __expf(M[r] - Mn);
            M[r] = Mn;
            rsum[r] = 0.0f;
        }

        // ---- P = sgn(s)*exp(|s| - M), row sum of |P| ----
#pragma unroll
        for (int t = 0; t < 8; ++t)
#pragma unroll
            for (int r = 0; r < 8; ++r) {
                float sv = sacc[t][r];
                float p = 0.0f;
                if (sv != 0.0f) {
                    float e = __expf(fabsf(sv) - M[r]);
                    p = (sv > 0.0f) ? e : -e;
                    rsum[r] += e;
                }
                sacc[t][r] = p;
            }
#pragma unroll
        for (int m = 1; m < 16; m <<= 1)
#pragma unroll
            for (int r = 0; r < 8; ++r)
                rsum[r] += __shfl_xor(rsum[r], m, 32);
#pragma unroll
        for (int r = 0; r < 8; ++r) L[r] = L[r] * alpha[r] + rsum[r];
#pragma unroll
        for (int t = 0; t < 8; ++t)
#pragma unroll
            for (int r = 0; r < 8; ++r) zacc[t][r] *= alpha[r];

        // ---- P -> per-wave LDS slab (cross-lane transpose for A frags) ----
#pragma unroll
        for (int t = 0; t < 8; ++t)
#pragma unroll
            for (int r = 0; r < 8; ++r)
                pw[(8 * lh + r) * KS + t * 16 + ln] = (_Float16)sacc[t][r];
        wait_lds();

        h16 ap[4];
#pragma unroll
        for (int c = 0; c < 4; ++c) {
            HU u;
            u.h[0] = *(const h8*)(pw + ln * KS + 32 * c + 8 * lh);
            u.h[1] = *(const h8*)(pw + ln * KS + 32 * c + 16 + 8 * lh);
            ap[c] = u.v;
        }

        // ---- Z += P * V ----
#pragma unroll
        for (int t2 = 0; t2 < 8; ++t2) {
            const _Float16* vp = Vl + (t2 * 16 + ln) * KS + 8 * lh;
#pragma unroll
            for (int c = 0; c < 4; ++c) {
                HU u;
                u.h[0] = *(const h8*)(vp + 32 * c);
                u.h[1] = *(const h8*)(vp + 32 * c + 16);
                zacc[t2] = wmma16(ap[c], u.v, zacc[t2]);
            }
        }
        __syncthreads();   // all waves done with Kl/Vl before restage
    }

    // ---- normalize and store z (f16, [s][h*128+d]) ----
    const int s0 = qb * 128 + w * 16 + 8 * lh;
#pragma unroll
    for (int t2 = 0; t2 < 8; ++t2)
#pragma unroll
        for (int r = 0; r < 8; ++r) {
            float vz = zacc[t2][r] / (L[r] + 1e-6f);
            zh[(size_t)(s0 + r) * DMODEL + head * DHEAD + t2 * 16 + ln] = (_Float16)vz;
        }
}

// ---------------------------------------------------------------------------
// Kernel 4: output projection  out = z(f16) * W_O + b_O   (f32 out)
// ---------------------------------------------------------------------------
__global__ __launch_bounds__(256) void out_gemm(
    const _Float16* __restrict__ zh, const float* __restrict__ Wo,
    const float* __restrict__ bo, float* __restrict__ out)
{
    const int bn = blockIdx.x, bm = blockIdx.y;
    __shared__ __align__(32) _Float16 Al[128 * GS];
    __shared__ __align__(32) _Float16 Bl[128 * GS];

    const int tid = threadIdx.x;
    const int w   = tid >> 5, lane = tid & 31;
    const int wr  = w >> 1,   wc   = w & 1;
    const int ln  = lane & 15, lh  = lane >> 4;

    f8 acc[2][4];
#pragma unroll
    for (int i = 0; i < 2; ++i)
#pragma unroll
        for (int t = 0; t < 4; ++t) acc[i][t] = f8zero();

    for (int kt = 0; kt < DMODEL / 32; ++kt) {
        const int k0 = kt * 32;
#pragma unroll
        for (int i = 0; i < 2; ++i) {
            int f = tid + 256 * i;          // 512 h8's
            int row = f >> 2, c8 = (f & 3) << 3;
            *(h8*)(Al + row * GS + c8) =
                *(const h8*)(zh + (size_t)(bm * 128 + row) * DMODEL + k0 + c8);
        }
#pragma unroll
        for (int i = 0; i < 4; ++i) {
            int f = tid + 256 * i;
            int k = f >> 5, n4 = (f & 31) << 2;
            f4 v = *(const f4*)(Wo + (size_t)(k0 + k) * DMODEL + bn * 128 + n4);
#pragma unroll
            for (int j = 0; j < 4; ++j) Bl[(n4 + j) * GS + k] = (_Float16)v[j];
        }
        if (kt < DMODEL / 32 - 1) {
            if (tid < 128)
                __builtin_prefetch(zh + (size_t)(bm * 128 + tid) * DMODEL + k0 + 32, 0, 1);
            else {
                int kr = (tid - 128) >> 2, q4 = (tid - 128) & 3;
                __builtin_prefetch(Wo + (size_t)(k0 + 32 + kr) * DMODEL +
                                   bn * 128 + q4 * 32, 0, 1);
            }
        }
        __syncthreads();

        h16 a[2], b[4];
#pragma unroll
        for (int i = 0; i < 2; ++i) {
            int row = wr * 32 + i * 16 + ln;
            HU u;
            u.h[0] = *(const h8*)(Al + row * GS + 8 * lh);
            u.h[1] = *(const h8*)(Al + row * GS + 16 + 8 * lh);
            a[i] = u.v;
        }
#pragma unroll
        for (int t = 0; t < 4; ++t) {
            int col = wc * 64 + t * 16 + ln;
            HU u;
            u.h[0] = *(const h8*)(Bl + col * GS + 8 * lh);
            u.h[1] = *(const h8*)(Bl + col * GS + 16 + 8 * lh);
            b[t] = u.v;
        }
#pragma unroll
        for (int i = 0; i < 2; ++i)
#pragma unroll
            for (int t = 0; t < 4; ++t)
                acc[i][t] = wmma16(a[i], b[t], acc[i][t]);
        __syncthreads();
    }

#pragma unroll
    for (int i = 0; i < 2; ++i)
#pragma unroll
        for (int t = 0; t < 4; ++t) {
            int col = bn * 128 + wc * 64 + t * 16 + ln;
            float bb = bo[col];
            int s0 = bm * 128 + wr * 32 + i * 16 + 8 * lh;
#pragma unroll
            for (int r = 0; r < 8; ++r)
                out[(size_t)(s0 + r) * DMODEL + col] = acc[i][t][r] + bb;
        }
}

// ---------------------------------------------------------------------------
extern "C" void kernel_launch(void* const* d_in, const int* in_sizes, int n_in,
                              void* d_out, int out_size, void* d_ws, size_t ws_size,
                              hipStream_t stream)
{
    const float* resid = (const float*)d_in[0];
    const float* Wq    = (const float*)d_in[1];
    const float* Wk    = (const float*)d_in[2];
    const float* Wv    = (const float*)d_in[3];
    const float* Wo    = (const float*)d_in[4];
    const float* bq    = (const float*)d_in[5];
    const float* bk    = (const float*)d_in[6];
    const float* bv    = (const float*)d_in[7];
    const float* bo    = (const float*)d_in[8];

    char* ws = (char*)d_ws;
    const size_t MB = 1024 * 1024;
    float*    qraw = (float*)(ws);                       // 16 MB
    float*    kraw = (float*)(ws + 16 * MB);             // 16 MB
    _Float16* qh   = (_Float16*)(ws + 32 * MB);          //  8 MB
    _Float16* kh   = (_Float16*)(ws + 40 * MB);          //  8 MB
    _Float16* vt   = (_Float16*)(ws + 48 * MB);          //  8 MB
    _Float16* zh   = (_Float16*)(ws + 56 * MB);          //  8 MB  (total 64 MB)

    proj_gemm<0><<<dim3(16, 16), 256, 0, stream>>>(resid, Wq, bq, qraw, nullptr);
    proj_gemm<0><<<dim3(16, 16), 256, 0, stream>>>(resid, Wk, bk, kraw, nullptr);
    proj_gemm<1><<<dim3(16, 16), 256, 0, stream>>>(resid, Wv, bv, nullptr, vt);
    rope_kernel<<<(SEQ * 1024) / 256, 256, 0, stream>>>(qraw, kraw, qh, kh);

    const size_t attn_lds = (size_t)(2 * 128 * KS + 8 * 16 * KS) * sizeof(_Float16);
    attn_kernel<<<dim3(16, 16), 256, attn_lds, stream>>>(qh, kh, vt, zh);

    out_gemm<<<dim3(16, 16), 256, 0, stream>>>(zh, Wo, bo, (float*)d_out);
}
